// FFTSelfAttention_6399501271425
// MI455X (gfx1250) — matrix operand order repaired and checked by reading, not compiled, not run on hardware
//
#include <hip/hip_runtime.h>

// ---------------------------------------------------------------------------
// FFT self-attention (circular-correlation attention) for MI455X / gfx1250.
// 4 x [16384,1024]x[1024,1024]^T bf16 WMMA GEMMs + tiny fp32 correlation.
// Global->LDS staging uses CDNA5 async-to-LDS (ASYNCcnt) instructions.
// ---------------------------------------------------------------------------

typedef __attribute__((ext_vector_type(16))) __bf16 v16bf;
typedef __attribute__((ext_vector_type(8)))  __bf16 v8bf;
typedef __attribute__((ext_vector_type(8)))  float  v8f;

#define DIM       1024
#define NUM_HEADS 16
#define HD        64
#define BATCH     4
#define SEQ       4096
#define M_TOTAL   (BATCH * SEQ)   // 16384 tokens

// GEMM tiling
#define BM  128
#define BN  128
#define BK  32
#define LDT 40                    // padded LDS row stride (elements), 80B = 5*16B

__device__ __forceinline__ __bf16 f2bf(float f) {
  union { float f; unsigned u; } v; v.f = f;
  unsigned r = v.u + 0x7FFFu + ((v.u >> 16) & 1u);   // round-to-nearest-even
  unsigned short h = (unsigned short)(r >> 16);
  __bf16 o;
  __builtin_memcpy(&o, &h, 2);
  return o;
}

__device__ __forceinline__ v16bf cat8(v8bf lo, v8bf hi) {
  v16bf r;
#pragma unroll
  for (int i = 0; i < 8; ++i) { r[i] = lo[i]; r[i + 8] = hi[i]; }
  return r;
}

// 32-byte global -> LDS async copy per lane (2 x ASYNC_TO_LDS_B128).
// INST_OFFSET applies to both LDS and memory addresses (ISA 15.18.3), so one
// address pair covers both 16B beats.
__device__ __forceinline__ void async_copy32(const __bf16* g, __bf16* l) {
  unsigned lds = (unsigned)(size_t)l;                 // low 32b = LDS offset
  unsigned long long ga = (unsigned long long)(size_t)g;
  asm volatile("global_load_async_to_lds_b128 %0, %1, off"
               :: "v"(lds), "v"(ga) : "memory");
  asm volatile("global_load_async_to_lds_b128 %0, %1, off offset:16"
               :: "v"(lds), "v"(ga) : "memory");
}

__device__ __forceinline__ void wait_async0() {
  asm volatile("s_wait_asynccnt 0x0" ::: "memory");
}

// ---------------------------------------------------------------------------
// fp32 -> bf16 conversion, 8 elements / thread (n must be a multiple of 8)
// ---------------------------------------------------------------------------
__global__ void __launch_bounds__(256)
cvt_f32_bf16(const float* __restrict__ s, __bf16* __restrict__ d, int n8) {
  int i = blockIdx.x * blockDim.x + threadIdx.x;
  if (i >= n8) return;
  const float4* sp = (const float4*)s + (size_t)i * 2;
  float4 a = sp[0], b = sp[1];
  v8bf o;
  o[0] = f2bf(a.x); o[1] = f2bf(a.y); o[2] = f2bf(a.z); o[3] = f2bf(a.w);
  o[4] = f2bf(b.x); o[5] = f2bf(b.y); o[6] = f2bf(b.z); o[7] = f2bf(b.w);
  ((v8bf*)d)[i] = o;
}

// ---------------------------------------------------------------------------
// C[M,1024] = A[M,1024](bf16) x W[1024,1024](bf16, row e = output feature,
// K contiguous)^T + bias.  Block: 128x128 tile, 256 threads (8 waves).
// Each wave: 32(M) x 64(N) via 2x4 grid of 16x16x32 bf16 WMMAs.
// Double-buffered LDS fed by async-to-LDS copies (ASYNCcnt pipeline).
// ---------------------------------------------------------------------------
__global__ void __launch_bounds__(256)
gemm_bf16_wmma(const __bf16* __restrict__ A, const __bf16* __restrict__ W,
               const float* __restrict__ bias, float* __restrict__ C) {
  __shared__ __bf16 sA[2][BM * LDT];
  __shared__ __bf16 sB[2][BN * LDT];

  const int tid   = threadIdx.x;
  const int lane  = tid & 31;
  const int wave  = tid >> 5;
  const int waveM = wave >> 1;          // 0..3 -> M offset waveM*32
  const int waveN = wave & 1;           // 0..1 -> N offset waveN*64
  const int blockM = blockIdx.x * BM;
  const int blockN = blockIdx.y * BN;

  const int r = lane & 15;              // row/col within 16x16 tile
  const int h = lane >> 4;              // lane half selects K sub-block

  // Global->LDS staging: 2 threads per row, 32B per thread per matrix.
  const int grow = tid >> 1;            // 0..127
  const int gcol = (tid & 1) * 16;      // element offset in K tile
  const __bf16* gA = A + (size_t)(blockM + grow) * DIM + gcol;
  const __bf16* gB = W + (size_t)(blockN + grow) * DIM + gcol;
  const int sOff = grow * LDT + gcol;   // LDS element offset for this thread

  v8f acc[2][4];
#pragma unroll
  for (int mi = 0; mi < 2; ++mi)
#pragma unroll
    for (int ni = 0; ni < 4; ++ni)
      acc[mi][ni] = (v8f){0.f, 0.f, 0.f, 0.f, 0.f, 0.f, 0.f, 0.f};

  // Prologue: async-fill LDS buffer 0 with K-tile 0.
  async_copy32(gA, &sA[0][sOff]);
  async_copy32(gB, &sB[0][sOff]);
  wait_async0();
  __syncthreads();

  int buf = 0;
#pragma unroll 2
  for (int k0 = 0; k0 < DIM; k0 += BK) {
    const bool more = (k0 + BK) < DIM;
    if (more) {                           // async prefetch next K tile
      const int nxt = buf ^ 1;            // safe: last read before prev barrier
      async_copy32(gA + (k0 + BK), &sA[nxt][sOff]);
      async_copy32(gB + (k0 + BK), &sB[nxt][sOff]);
    }

    // A fragments: 16x32 bf16, lane half h holds K = h*8..h*8+7 (v0..3)
    // and K = h*8+16..h*8+23 (v4..7).
    v16bf afrag[2];
#pragma unroll
    for (int mi = 0; mi < 2; ++mi) {
      const __bf16* p = &sA[buf][(waveM * 32 + mi * 16 + r) * LDT + h * 8];
      afrag[mi] = cat8(*(const v8bf*)p, *(const v8bf*)(p + 16));
    }
    // B fragments: 32x16 bf16, lane half h holds K = h*8..h*8+7 and h*8+16..h*8+23.
    v16bf bfrag[4];
#pragma unroll
    for (int ni = 0; ni < 4; ++ni) {
      const __bf16* p = &sB[buf][(waveN * 64 + ni * 16 + r) * LDT + h * 8];
      bfrag[ni] = cat8(*(const v8bf*)p, *(const v8bf*)(p + 16));
    }

#pragma unroll
    for (int mi = 0; mi < 2; ++mi)
#pragma unroll
      for (int ni = 0; ni < 4; ++ni)
        acc[mi][ni] = __builtin_amdgcn_wmma_f32_16x16x32_bf16(
            false, afrag[mi], false, bfrag[ni], (short)0, acc[mi][ni],
            false, false);

    if (more) {
      wait_async0();     // my async writes into buf^1 have landed in LDS
      __syncthreads();   // everyone's landed; everyone done reading buf
      buf ^= 1;
    }
  }

  // Store: C/D layout -> VGPR i of lane holds (M = h*8 + i, N = r).
#pragma unroll
  for (int mi = 0; mi < 2; ++mi) {
#pragma unroll
    for (int ni = 0; ni < 4; ++ni) {
      const int gN = blockN + waveN * 64 + ni * 16 + r;
      const float bv = bias[gN];
      const int rowBase = blockM + waveM * 32 + mi * 16 + h * 8;
      v8f c = acc[mi][ni];
#pragma unroll
      for (int i = 0; i < 8; ++i)
        C[(size_t)(rowBase + i) * DIM + gN] = c[i] + bv;
    }
  }
}

// ---------------------------------------------------------------------------
// attn[n] = sum_j q[j] * k[(j-n) & 63]; out = attn * v, emitted as bf16.
// Block: 256 threads = 4 (token,head) pairs x 64 outputs.
// ---------------------------------------------------------------------------
__global__ void __launch_bounds__(256)
corr_gate(const float* __restrict__ q, const float* __restrict__ k,
          const float* __restrict__ v, __bf16* __restrict__ out) {
  __shared__ float sq[4][HD];
  __shared__ float sk[4][HD];
  const int tid = threadIdx.x;
  const int n   = tid & 63;
  const int hh  = tid >> 6;                       // local (token,head) slot
  const int flat = blockIdx.x * 4 + hh;           // token*NUM_HEADS + head
  const int token = flat >> 4;
  const int head  = flat & 15;
  const size_t base = (size_t)token * DIM + head * HD;

  sq[hh][n] = q[base + n];
  sk[hh][n] = k[base + n];
  __syncthreads();

  float acc = 0.f;
#pragma unroll
  for (int j = 0; j < HD; ++j)
    acc += sq[hh][j] * sk[hh][(j - n) & 63];

  out[base + n] = f2bf(acc * v[base + n]);
}

// ---------------------------------------------------------------------------
extern "C" void kernel_launch(void* const* d_in, const int* in_sizes, int n_in,
                              void* d_out, int out_size, void* d_ws,
                              size_t ws_size, hipStream_t stream) {
  const float* x  = (const float*)d_in[0];
  const float* Wq = (const float*)d_in[1];
  const float* bq = (const float*)d_in[2];
  const float* Wk = (const float*)d_in[3];
  const float* bk = (const float*)d_in[4];
  const float* Wv = (const float*)d_in[5];
  const float* bv = (const float*)d_in[6];
  const float* Wo = (const float*)d_in[7];
  const float* bo = (const float*)d_in[8];
  float* out = (float*)d_out;

  // Workspace carve-up.
  char* p = (char*)d_ws;
  __bf16* xbf   = (__bf16*)p; p += (size_t)M_TOTAL * DIM * 2;   // 32 MB
  __bf16* wqb   = (__bf16*)p; p += (size_t)DIM * DIM * 2;       //  2 MB
  __bf16* wkb   = (__bf16*)p; p += (size_t)DIM * DIM * 2;
  __bf16* wvb   = (__bf16*)p; p += (size_t)DIM * DIM * 2;
  __bf16* wob   = (__bf16*)p; p += (size_t)DIM * DIM * 2;
  float*  qf    = (float*)p;  p += (size_t)M_TOTAL * DIM * 4;   // 64 MB
  float*  kf    = (float*)p;  p += (size_t)M_TOTAL * DIM * 4;
  float*  vf    = (float*)p;  p += (size_t)M_TOTAL * DIM * 4;
  __bf16* gated = (__bf16*)p; p += (size_t)M_TOTAL * DIM * 2;   // 32 MB

  // 1) fp32 -> bf16 conversions.
  {
    int n8 = (M_TOTAL * DIM) / 8;
    cvt_f32_bf16<<<(n8 + 255) / 256, 256, 0, stream>>>(x, xbf, n8);
    int w8 = (DIM * DIM) / 8;
    cvt_f32_bf16<<<(w8 + 255) / 256, 256, 0, stream>>>(Wq, wqb, w8);
    cvt_f32_bf16<<<(w8 + 255) / 256, 256, 0, stream>>>(Wk, wkb, w8);
    cvt_f32_bf16<<<(w8 + 255) / 256, 256, 0, stream>>>(Wv, wvb, w8);
    cvt_f32_bf16<<<(w8 + 255) / 256, 256, 0, stream>>>(Wo, wob, w8);
  }

  // 2) Q/K/V projections (WMMA bf16, f32 accumulate + bias).
  dim3 grid(M_TOTAL / BM, DIM / BN);   // 128 x 8
  gemm_bf16_wmma<<<grid, 256, 0, stream>>>(xbf, wqb, bq, qf);
  gemm_bf16_wmma<<<grid, 256, 0, stream>>>(xbf, wkb, bk, kf);
  gemm_bf16_wmma<<<grid, 256, 0, stream>>>(xbf, wvb, bv, vf);

  // 3) Circular cross-correlation + V gating (fp32 VALU, bf16 out).
  corr_gate<<<(M_TOTAL * NUM_HEADS) / 4, 256, 0, stream>>>(qf, kf, vf, gated);

  // 4) Output projection.
  gemm_bf16_wmma<<<grid, 256, 0, stream>>>(gated, wob, bo, out);
}
